// SparseCrosscoder_65584150610058
// MI455X (gfx1250) — matrix operand order repaired and checked by reading, not compiled, not run on hardware
//
#include <hip/hip_runtime.h>
#include <hip/hip_bf16.h>

// Problem sizes (match reference)
#define BATCH 4096
#define LAYERS 4
#define DMODEL 1024
#define HID 16384
#define KDIM (LAYERS * DMODEL)   // 4096 reduction dim
#define TOPK 64

// GEMM tiling: block tile 128(M) x 256(N), 8 waves as 2(M) x 4(N), wave tile 64x64
#define TM 128
#define TN 256
#define TK 32
#define LDSTR 40                  // bf16 elems per LDS row (32 + 8 pad), 80B, 16B-aligned

typedef __bf16 bf16_t;
typedef __attribute__((ext_vector_type(16))) __bf16 v16bf;
typedef __attribute__((ext_vector_type(8)))  __bf16 v8bf;
typedef __attribute__((ext_vector_type(8)))  float  v8f;

union AFrag { v16bf v; v8bf h[2]; };

// ---------------------------------------------------------------------------
// Kernel 1: h = (concat(x) - b_pre) @ W_enc + b_enc, written into f region.
// bf16 WMMA (v_wmma_f32_16x16x32_bf16), f32 accumulate.
// ---------------------------------------------------------------------------
__global__ __launch_bounds__(256) void enc_gemm(
    const float* __restrict__ x0, const float* __restrict__ x1,
    const float* __restrict__ x2, const float* __restrict__ x3,
    const float* __restrict__ b_pre, const float* __restrict__ W_enc,
    const float* __restrict__ b_enc, float* __restrict__ h_out)
{
    __shared__ __align__(16) bf16_t As[TM * LDSTR];  // [m][k], k-contiguous
    __shared__ __align__(16) bf16_t Bs[TN * LDSTR];  // [n][k], k-contiguous (B transposed)

    const int tid  = threadIdx.x;
    const int lane = tid & 31;
    const int wave = tid >> 5;
    const int wm   = wave & 1;     // 2 wave-rows of 64
    const int wn   = wave >> 1;    // 4 wave-cols of 64
    // blockIdx.x walks M (x fastest): concurrent blocks share the W_enc slab,
    // and the operand that gets re-read across N-slabs is x (64 MB, L2-resident).
    const int m0   = blockIdx.x * TM;
    const int n0   = blockIdx.y * TN;

    const float* xs[LAYERS] = {x0, x1, x2, x3};

    v8f acc[4][4];
    for (int i = 0; i < 4; ++i)
        for (int j = 0; j < 4; ++j)
            for (int e = 0; e < 8; ++e) acc[i][j][e] = 0.0f;

    const int nlo = lane & 15;
    const int kbA = (lane < 16) ? 0 : 8;    // A lane K-base (16-bit A 16x32 layout)
    const int kbB = (lane < 16) ? 0 : 8;    // B lane K-base (16-bit B 32x16 layout)

    for (int k0 = 0; k0 < KDIM; k0 += TK) {
        const int layer = k0 >> 10;         // TK=32 never crosses a 1024 boundary
        const int koff  = k0 & 1023;
        const float* __restrict__ xl = xs[layer];

        // --- stage A tile: 128x32 f32 -> bf16 LDS. 8 threads/row, 4 floats each.
        {
            const int c4    = (tid & 7) * 4;
            const int rbase = tid >> 3;       // 0..31
            const float4 bp = *(const float4*)(b_pre + k0 + c4);
            for (int rr = 0; rr < TM; rr += 32) {
                const int r = rbase + rr;
                const float4 v = *(const float4*)(xl + (size_t)(m0 + r) * DMODEL + koff + c4);
                bf16_t* dst = &As[r * LDSTR + c4];
                dst[0] = (bf16_t)(v.x - bp.x);
                dst[1] = (bf16_t)(v.y - bp.y);
                dst[2] = (bf16_t)(v.z - bp.z);
                dst[3] = (bf16_t)(v.w - bp.w);
            }
        }
        // --- stage B tile: W_enc[k0..k0+31][n0..n0+255] -> Bs[n][k] (transposed)
        {
            const int n4 = (tid & 63) * 4;    // 64 col-groups of 4 floats = 256 cols
            const int kr = tid >> 6;          // 0..3
            for (int kk = 0; kk < TK; kk += 4) {
                const int krow = kr + kk;
                const float* src = W_enc + (size_t)(k0 + krow) * HID + n0 + n4;
                const float4 v = *(const float4*)src;
                Bs[(n4 + 0) * LDSTR + krow] = (bf16_t)v.x;
                Bs[(n4 + 1) * LDSTR + krow] = (bf16_t)v.y;
                Bs[(n4 + 2) * LDSTR + krow] = (bf16_t)v.z;
                Bs[(n4 + 3) * LDSTR + krow] = (bf16_t)v.w;
                // prefetch next K-step's B tile (global_prefetch_b8)
                if (k0 + TK < KDIM)
                    __builtin_prefetch(src + (size_t)TK * HID, 0, 0);
            }
        }
        __syncthreads();

        // --- build fragments per ISA lane layouts, 2x ds_load_b128 each
        AFrag b[4];
        #pragma unroll
        for (int j = 0; j < 4; ++j) {
            const int n = wn * 64 + j * 16 + nlo;
            b[j].h[0] = *(const v8bf*)&Bs[n * LDSTR + kbB];        // K = kbB .. kbB+7
            b[j].h[1] = *(const v8bf*)&Bs[n * LDSTR + kbB + 16];   // K = kbB+16 .. kbB+23
        }
        #pragma unroll
        for (int i = 0; i < 4; ++i) {
            AFrag a;
            const int m = wm * 64 + i * 16 + nlo;
            a.h[0] = *(const v8bf*)&As[m * LDSTR + kbA];           // K = kbA .. kbA+7
            a.h[1] = *(const v8bf*)&As[m * LDSTR + kbA + 16];      // K = kbA+16 .. kbA+23
            #pragma unroll
            for (int j = 0; j < 4; ++j)
                acc[i][j] = __builtin_amdgcn_wmma_f32_16x16x32_bf16(
                    false, a.v, false, b[j].v, (short)0, acc[i][j], false, false);
        }

        __syncthreads();
    }

    // --- epilogue: + b_enc, write h into f region of d_out
    // C/D layout: VGPR e -> M = e + 8*(lane>=16), N = lane&15
    const int mup = (lane >> 4) << 3;
    for (int i = 0; i < 4; ++i) {
        const int rbase = m0 + wm * 64 + i * 16 + mup;
        for (int j = 0; j < 4; ++j) {
            const int col = n0 + wn * 64 + j * 16 + nlo;
            const float be = b_enc[col];
            for (int e = 0; e < 8; ++e)
                h_out[(size_t)(rbase + e) * HID + col] = acc[i][j][e] + be;
        }
    }
}

// ---------------------------------------------------------------------------
// Kernel 2: per-row top-K radix select, in place on the f region.
// ---------------------------------------------------------------------------
__device__ inline unsigned tokey(float f) {
    unsigned u = __float_as_uint(f);
    return (u & 0x80000000u) ? ~u : (u | 0x80000000u);  // monotone: larger float -> larger key
}

__global__ __launch_bounds__(256) void topk_select(
    float* __restrict__ f, float* __restrict__ tv, int* __restrict__ ti)
{
    const int row = blockIdx.x;
    float* __restrict__ frow = f + (size_t)row * HID;
    const int tid = threadIdx.x;

    __shared__ unsigned hist[256];
    __shared__ unsigned s_chosen, s_want;
    __shared__ int s_tie, s_nsel;

    unsigned prefix = 0;
    unsigned want   = TOPK;         // rank among descending keys

    for (int byte = 3; byte >= 0; --byte) {
        const int shift = byte * 8;
        hist[tid] = 0;
        __syncthreads();
        for (int e = tid; e < HID; e += 256) {
            const unsigned key = tokey(frow[e]);
            const bool match = (byte == 3) || (((key ^ prefix) >> (shift + 8)) == 0);
            if (match) atomicAdd(&hist[(key >> shift) & 255], 1u);
        }
        __syncthreads();
        if (tid == 0) {
            unsigned cum = 0, d = 0, w = want;
            for (int dd = 255; dd >= 0; --dd) {
                const unsigned c = hist[dd];
                if (cum + c >= w) { d = (unsigned)dd; w -= cum; break; }
                cum += c;
            }
            s_chosen = d; s_want = w;
        }
        __syncthreads();
        prefix |= (s_chosen << shift);
        want = s_want;
        __syncthreads();            // hist reused next pass
    }

    if (tid == 0) { s_tie = 0; s_nsel = 0; }
    __syncthreads();
    const unsigned T = prefix;      // key of K-th largest
    const int quota  = (int)want;   // how many == T to keep

    for (int e = tid; e < HID; e += 256) {
        const float v = frow[e];
        const unsigned key = tokey(v);
        bool keep = (key > T);
        if (!keep && key == T) keep = (atomicAdd(&s_tie, 1) < quota);
        float out = 0.0f;
        if (keep) {
            out = v > 0.0f ? v : 0.0f;               // relu(topk_vals)
            const int slot = atomicAdd(&s_nsel, 1);
            if (slot < TOPK) { tv[row * TOPK + slot] = out; ti[row * TOPK + slot] = e; }
        }
        frow[e] = out;
    }
}

// ---------------------------------------------------------------------------
// Kernel 3: sparse decode. recons[l][b][:] = b_dec[l] + sum_k val * W_dec[l,:,idx]
// ---------------------------------------------------------------------------
__global__ __launch_bounds__(256) void decode(
    const float* __restrict__ tv, const int* __restrict__ ti,
    const float* __restrict__ W_dec, const float* __restrict__ b_dec,
    float* __restrict__ out)
{
    const int b = blockIdx.x;
    const int l = blockIdx.y;
    const int tid = threadIdx.x;

    __shared__ float sv[TOPK];
    __shared__ int   si[TOPK];
    if (tid < TOPK) { sv[tid] = tv[b * TOPK + tid]; si[tid] = ti[b * TOPK + tid]; }
    __syncthreads();

    const float* __restrict__ Wl = W_dec + (size_t)l * DMODEL * HID;
    for (int d = tid; d < DMODEL; d += 256) {
        float acc = b_dec[l * DMODEL + d];
        const float* __restrict__ wrow = Wl + (size_t)d * HID;
        #pragma unroll 8
        for (int k = 0; k < TOPK; ++k)
            acc = fmaf(sv[k], wrow[si[k]], acc);
        out[((size_t)l * BATCH + b) * DMODEL + d] = acc;
    }
}

// ---------------------------------------------------------------------------
extern "C" void kernel_launch(void* const* d_in, const int* in_sizes, int n_in,
                              void* d_out, int out_size, void* d_ws, size_t ws_size,
                              hipStream_t stream)
{
    const float* x0    = (const float*)d_in[0];
    const float* x1    = (const float*)d_in[1];
    const float* x2    = (const float*)d_in[2];
    const float* x3    = (const float*)d_in[3];
    const float* b_pre = (const float*)d_in[4];
    const float* W_enc = (const float*)d_in[5];
    const float* b_enc = (const float*)d_in[6];
    const float* W_dec = (const float*)d_in[7];
    const float* b_dec = (const float*)d_in[8];

    float* out    = (float*)d_out;
    float* recons = out;                                       // 4 * [B, D]
    float* f      = out + (size_t)LAYERS * BATCH * DMODEL;     // [B, H]

    float* tv = (float*)d_ws;                                  // [B, K] relu'd vals
    int*   ti = (int*)((char*)d_ws + (size_t)BATCH * TOPK * sizeof(float));

    dim3 g1(BATCH / TM, HID / TN);   // x = M (32), y = N (64): W_enc streamed once
    enc_gemm<<<g1, 256, 0, stream>>>(x0, x1, x2, x3, b_pre, W_enc, b_enc, f);

    topk_select<<<BATCH, 256, 0, stream>>>(f, tv, ti);

    dim3 g3(BATCH, LAYERS);
    decode<<<g3, 256, 0, stream>>>(tv, ti, W_dec, b_dec, recons);
}
